// Model_57501022159430
// MI455X (gfx1250) — matrix operand order repaired and checked by reading, not compiled, not run on hardware
//
#include <hip/hip_runtime.h>

// ---------------- problem constants ----------------
#define D_   1024
#define N_   32
#define T_   750
#define C_   20
#define M_   (N_ * T_)          // 24000
#define KSEL_ 657               // K = T - T//S = 750 - 93
#define OMEGA_ 0.6f
#define MASKV_ (-100.0f)

// output layout (flat float elements, reference return order)
constexpr size_t OFF_XR   = 0;
constexpr size_t OFF_CXR  = (size_t)M_ * D_;                 // 24,576,000
constexpr size_t OFF_CRA  = OFF_CXR + (size_t)M_ * C_;       // 25,056,000
constexpr size_t OFF_XF   = OFF_CRA + (size_t)M_ * C_;       // 25,536,000
constexpr size_t OFF_CXF  = OFF_XF  + (size_t)M_ * D_;       // 50,112,000
constexpr size_t OFF_CFA  = OFF_CXF + (size_t)M_ * C_;       // 50,592,000
constexpr size_t OFF_TCAM = OFF_CFA + (size_t)M_ * C_;       // 51,072,000

// ---------------- WMMA plumbing ----------------
typedef __attribute__((ext_vector_type(16))) __bf16 v16bf;
typedef __attribute__((ext_vector_type(8)))  __bf16 v8bf;
typedef __attribute__((ext_vector_type(8)))  float  v8f;
typedef __attribute__((ext_vector_type(4)))  float  v4f;

union FragBF { v16bf v; v8bf h[2]; };

__device__ __forceinline__ v8f wmma_bf16(const FragBF& a, const FragBF& b, v8f c) {
  return __builtin_amdgcn_wmma_f32_16x16x32_bf16(false, a.v, false, b.v,
                                                 (short)0, c, false, false);
}

// A fragment: 16x32 bf16. lane<16 holds row M=lane, K in [0,8)+[16,24);
// lane>=16 holds row M=lane-16, K in [8,16)+[24,32). LDS tile row-major stride 32.
__device__ __forceinline__ void frag_a(FragBF& f, const __bf16* Tl, int mBase, int lane) {
  const int half = lane >> 4;
  const __bf16* row = Tl + (size_t)(mBase + (lane & 15)) * 32;
  f.h[0] = *(const v8bf*)(row + half * 8);
  f.h[1] = *(const v8bf*)(row + 16 + half * 8);
}
// B fragment: 32x16 bf16, column N=lane%16; lane<16 holds K in [0,8)+[16,24), lane>=16 K in [8,16)+[24,32).
// We store B tile as W rows (row-major stride 32), i.e. column n of B == row n of W.
__device__ __forceinline__ void frag_b(FragBF& f, const __bf16* Tl, int nBase, int lane) {
  const int half = lane >> 4;
  const __bf16* row = Tl + (size_t)(nBase + (lane & 15)) * 32;
  f.h[0] = *(const v8bf*)(row + half * 8);
  f.h[1] = *(const v8bf*)(row + 16 + half * 8);
}

__device__ __forceinline__ void cvt8(__bf16* d, const float* src) {
  #pragma unroll
  for (int i = 0; i < 8; i += 4) {
    v4f f = *(const v4f*)(src + i);
    d[i+0] = (__bf16)f[0]; d[i+1] = (__bf16)f[1];
    d[i+2] = (__bf16)f[2]; d[i+3] = (__bf16)f[3];
  }
}

// ---------------- GEMM1: h1 = relu(X @ W^T + b), out bf16 ----------------
// Block tile 64x256, 8 waves (2 M x 4 N), wave tile 32x64: 8 WMMA / 6 frags per K-step.
__global__ __launch_bounds__(256) void k_gemm1(
    const float* __restrict__ X,
    const float* __restrict__ Wr, const float* __restrict__ br,
    const float* __restrict__ Wf, const float* __restrict__ bfl,
    __bf16* __restrict__ H1)
{
  __shared__ __bf16 As[64 * 32];
  __shared__ __bf16 Bs[256 * 32];
  const int s  = blockIdx.z;
  const float* W    = s ? Wf  : Wr;
  const float* bias = s ? bfl : br;
  const int m0 = blockIdx.x * 64;
  const int n0 = blockIdx.y * 256;
  const int tid = threadIdx.x, lane = tid & 31, wid = tid >> 5;
  const int waveM = wid & 1, waveN = wid >> 1;   // 2 x 4 waves, each 32x64
  const int arow = tid >> 2, acs = (tid & 3) * 8;

  v8f zero = {0,0,0,0,0,0,0,0};
  v8f acc[2][4];
  #pragma unroll
  for (int i = 0; i < 2; ++i)
    #pragma unroll
    for (int j = 0; j < 4; ++j) acc[i][j] = zero;

  for (int k0 = 0; k0 < D_; k0 += 32) {
    __syncthreads();
    // A tile 64x32: 8 elems/thread
    cvt8(&As[arow * 32 + acs],
         X + (size_t)(m0 + arow) * (2 * D_) + (size_t)s * D_ + k0 + acs);
    // B tile 256x32: one row per thread (32 elems)
    {
      const float* src = W + (size_t)(n0 + tid) * D_ + k0;
      __bf16* d = &Bs[tid * 32];
      cvt8(d,      src);
      cvt8(d + 8,  src + 8);
      cvt8(d + 16, src + 16);
      cvt8(d + 24, src + 24);
    }
    __syncthreads();
    FragBF a0, a1, b0, b1, b2, b3;
    frag_a(a0, As, waveM * 32,      lane);
    frag_a(a1, As, waveM * 32 + 16, lane);
    frag_b(b0, Bs, waveN * 64,      lane);
    frag_b(b1, Bs, waveN * 64 + 16, lane);
    frag_b(b2, Bs, waveN * 64 + 32, lane);
    frag_b(b3, Bs, waveN * 64 + 48, lane);
    acc[0][0] = wmma_bf16(a0, b0, acc[0][0]);
    acc[0][1] = wmma_bf16(a0, b1, acc[0][1]);
    acc[0][2] = wmma_bf16(a0, b2, acc[0][2]);
    acc[0][3] = wmma_bf16(a0, b3, acc[0][3]);
    acc[1][0] = wmma_bf16(a1, b0, acc[1][0]);
    acc[1][1] = wmma_bf16(a1, b1, acc[1][1]);
    acc[1][2] = wmma_bf16(a1, b2, acc[1][2]);
    acc[1][3] = wmma_bf16(a1, b3, acc[1][3]);
  }

  const int half = lane >> 4, colL = lane & 15;
  __bf16* out = H1 + (size_t)s * M_ * D_;
  #pragma unroll
  for (int tm = 0; tm < 2; ++tm) {
    const int grow0 = m0 + waveM * 32 + tm * 16 + half * 8;
    #pragma unroll
    for (int tn = 0; tn < 4; ++tn) {
      const int gcol = n0 + waveN * 64 + tn * 16 + colL;
      const float bv = bias[gcol];
      #pragma unroll
      for (int r = 0; r < 8; ++r) {
        float v = acc[tm][tn][r] + bv;
        v = v > 0.f ? v : 0.f;
        out[(size_t)(grow0 + r) * D_ + gcol] = (__bf16)v;
      }
    }
  }
}

// ---------------- GEMM2: x = relu(h1 @ W1^T + b), out fp32 ----------------
__global__ __launch_bounds__(256) void k_gemm2(
    const __bf16* __restrict__ H1,
    const float* __restrict__ Wr, const float* __restrict__ br,
    const float* __restrict__ Wf, const float* __restrict__ bfl,
    float* __restrict__ OUT)
{
  __shared__ __bf16 As[64 * 32];
  __shared__ __bf16 Bs[256 * 32];
  const int s  = blockIdx.z;
  const float* W    = s ? Wf  : Wr;
  const float* bias = s ? bfl : br;
  const int m0 = blockIdx.x * 64;
  const int n0 = blockIdx.y * 256;
  const int tid = threadIdx.x, lane = tid & 31, wid = tid >> 5;
  const int waveM = wid & 1, waveN = wid >> 1;
  const int arow = tid >> 2, acs = (tid & 3) * 8;

  v8f zero = {0,0,0,0,0,0,0,0};
  v8f acc[2][4];
  #pragma unroll
  for (int i = 0; i < 2; ++i)
    #pragma unroll
    for (int j = 0; j < 4; ++j) acc[i][j] = zero;

  const __bf16* Hs = H1 + (size_t)s * M_ * D_;

  for (int k0 = 0; k0 < D_; k0 += 32) {
    __syncthreads();
    *(v8bf*)&As[arow * 32 + acs] =
        *(const v8bf*)(Hs + (size_t)(m0 + arow) * D_ + k0 + acs);
    {
      const float* src = W + (size_t)(n0 + tid) * D_ + k0;
      __bf16* d = &Bs[tid * 32];
      cvt8(d,      src);
      cvt8(d + 8,  src + 8);
      cvt8(d + 16, src + 16);
      cvt8(d + 24, src + 24);
    }
    __syncthreads();
    FragBF a0, a1, b0, b1, b2, b3;
    frag_a(a0, As, waveM * 32,      lane);
    frag_a(a1, As, waveM * 32 + 16, lane);
    frag_b(b0, Bs, waveN * 64,      lane);
    frag_b(b1, Bs, waveN * 64 + 16, lane);
    frag_b(b2, Bs, waveN * 64 + 32, lane);
    frag_b(b3, Bs, waveN * 64 + 48, lane);
    acc[0][0] = wmma_bf16(a0, b0, acc[0][0]);
    acc[0][1] = wmma_bf16(a0, b1, acc[0][1]);
    acc[0][2] = wmma_bf16(a0, b2, acc[0][2]);
    acc[0][3] = wmma_bf16(a0, b3, acc[0][3]);
    acc[1][0] = wmma_bf16(a1, b0, acc[1][0]);
    acc[1][1] = wmma_bf16(a1, b1, acc[1][1]);
    acc[1][2] = wmma_bf16(a1, b2, acc[1][2]);
    acc[1][3] = wmma_bf16(a1, b3, acc[1][3]);
  }

  const int half = lane >> 4, colL = lane & 15;
  float* out = OUT + (s ? OFF_XF : OFF_XR);
  #pragma unroll
  for (int tm = 0; tm < 2; ++tm) {
    const int grow0 = m0 + waveM * 32 + tm * 16 + half * 8;
    #pragma unroll
    for (int tn = 0; tn < 4; ++tn) {
      const int gcol = n0 + waveN * 64 + tn * 16 + colL;
      const float bv = bias[gcol];
      #pragma unroll
      for (int r = 0; r < 8; ++r) {
        float v = acc[tm][tn][r] + bv;
        out[(size_t)(grow0 + r) * D_ + gcol] = v > 0.f ? v : 0.f;
      }
    }
  }
}

// ---- GEMM3: cls_x = x @ Wcls^T + b ; cls_at = x @ Wa^T + b (40 cols, pad 64) ----
__global__ __launch_bounds__(256) void k_gemm3(
    const float* __restrict__ XIN,
    const float* __restrict__ Wcr, const float* __restrict__ bcr,
    const float* __restrict__ War, const float* __restrict__ bar_,
    const float* __restrict__ Wcf, const float* __restrict__ bcf,
    const float* __restrict__ Waf, const float* __restrict__ baf,
    float* __restrict__ OUT)
{
  __shared__ __bf16 As[64 * 32];
  __shared__ __bf16 Bs[64 * 32];
  const int s = blockIdx.z;
  const float* Wc = s ? Wcf : Wcr;
  const float* Wa = s ? Waf : War;
  const float* bc = s ? bcf : bcr;
  const float* ba = s ? baf : bar_;
  const float* Xs = XIN + (s ? OFF_XF : OFF_XR);
  float* cls_out  = OUT + (s ? OFF_CXF : OFF_CXR);
  float* at_out   = OUT + (s ? OFF_CFA : OFF_CRA);   // temporary: unmasked cls_at
  const int m0 = blockIdx.x * 64;
  const int tid = threadIdx.x, lane = tid & 31, wid = tid >> 5;
  const int waveM = wid & 1, waveN = wid >> 1;   // wave tile 32 rows x 16 cols
  const int arow = tid >> 2, acs = (tid & 3) * 8;

  v8f zero = {0,0,0,0,0,0,0,0};
  v8f acc[2] = {zero, zero};

  for (int k0 = 0; k0 < D_; k0 += 32) {
    __syncthreads();
    cvt8(&As[arow * 32 + acs], Xs + (size_t)(m0 + arow) * D_ + k0 + acs);
    {
      __bf16* d = &Bs[arow * 32 + acs];
      if (arow < C_) {
        cvt8(d, Wc + (size_t)arow * D_ + k0 + acs);
      } else if (arow < 2 * C_) {
        cvt8(d, Wa + (size_t)(arow - C_) * D_ + k0 + acs);
      } else {
        #pragma unroll
        for (int i = 0; i < 8; ++i) d[i] = (__bf16)0.f;
      }
    }
    __syncthreads();
    FragBF a0, a1, b0;
    frag_a(a0, As, waveM * 32,      lane);
    frag_a(a1, As, waveM * 32 + 16, lane);
    frag_b(b0, Bs, waveN * 16,      lane);
    acc[0] = wmma_bf16(a0, b0, acc[0]);
    acc[1] = wmma_bf16(a1, b0, acc[1]);
  }

  const int half = lane >> 4, colL = lane & 15;
  const int c = waveN * 16 + colL;
  #pragma unroll
  for (int tm = 0; tm < 2; ++tm) {
    const int grow0 = m0 + waveM * 32 + tm * 16 + half * 8;
    if (c < C_) {
      const float bv = bc[c];
      #pragma unroll
      for (int r = 0; r < 8; ++r)
        cls_out[(size_t)(grow0 + r) * C_ + c] = acc[tm][r] + bv;
    } else if (c < 2 * C_) {
      const int cc = c - C_;
      const float bv = ba[cc];
      #pragma unroll
      for (int r = 0; r < 8; ++r)
        at_out[(size_t)(grow0 + r) * C_ + cc] = acc[tm][r] + bv;
    }
  }
}

// ---- kth: stable-rank selection of sorted index KSEL_-1 over T per (s,n,c) ----
__global__ __launch_bounds__(256) void k_kth(
    const float* __restrict__ cls_r, const float* __restrict__ cls_f,
    float* __restrict__ kth)
{
  __shared__ float vals[T_];
  const int s = blockIdx.y;
  const int nc = blockIdx.x;            // n*C_ + c
  const int n = nc / C_, c = nc % C_;
  const float* src = (s ? cls_f : cls_r) + (size_t)n * T_ * C_ + c;
  for (int t = threadIdx.x; t < T_; t += blockDim.x) vals[t] = src[(size_t)t * C_];
  __syncthreads();
  for (int i = threadIdx.x; i < T_; i += blockDim.x) {
    const float v = vals[i];
    int rank = 0;
    for (int j = 0; j < T_; ++j) {
      const float u = vals[j];
      rank += (u < v) || (u == v && j < i);
    }
    if (rank == KSEL_ - 1) kth[(size_t)s * N_ * C_ + n * C_ + c] = v;
  }
}

// ---- final: mask cls_a slots in place, compute tcam ----
__global__ __launch_bounds__(256) void k_final(
    float* __restrict__ OUT, const float* __restrict__ kth,
    const float* __restrict__ mul_r, const float* __restrict__ mul_f)
{
  const int idx = blockIdx.x * blockDim.x + threadIdx.x;
  if (idx >= M_ * C_) return;
  const int c = idx % C_;
  const int n = idx / (T_ * C_);
  const float cxr = OUT[OFF_CXR + idx];
  const float atr = OUT[OFF_CRA + idx];   // unmasked cls_at_r (temp)
  const float cxf = OUT[OFF_CXF + idx];
  const float atf = OUT[OFF_CFA + idx];   // unmasked cls_at_f (temp)
  const float kr = kth[n * C_ + c];
  const float kf = kth[N_ * C_ + n * C_ + c];
  OUT[OFF_CRA + idx] = (cxr > kr) ? MASKV_ : atr;
  OUT[OFF_CFA + idx] = (cxf > kf) ? MASKV_ : atf;
  OUT[OFF_TCAM + idx] = (cxr + OMEGA_ * atr) * mul_r[c]
                      + (cxf + OMEGA_ * atf) * mul_f[c];
}

// ---------------- launch ----------------
extern "C" void kernel_launch(void* const* d_in, const int* in_sizes, int n_in,
                              void* d_out, int out_size, void* d_ws, size_t ws_size,
                              hipStream_t stream) {
  (void)in_sizes; (void)n_in; (void)out_size; (void)ws_size;
  const float* inputs  = (const float*)d_in[0];
  const float* W_fc_r  = (const float*)d_in[1];
  const float* b_fc_r  = (const float*)d_in[2];
  const float* W_fc1_r = (const float*)d_in[3];
  const float* b_fc1_r = (const float*)d_in[4];
  const float* W_fc_f  = (const float*)d_in[5];
  const float* b_fc_f  = (const float*)d_in[6];
  const float* W_fc1_f = (const float*)d_in[7];
  const float* b_fc1_f = (const float*)d_in[8];
  const float* W_cls_r = (const float*)d_in[9];
  const float* b_cls_r = (const float*)d_in[10];
  const float* W_cls_f = (const float*)d_in[11];
  const float* b_cls_f = (const float*)d_in[12];
  const float* W_ra    = (const float*)d_in[13];
  const float* b_ra    = (const float*)d_in[14];
  const float* W_fa    = (const float*)d_in[15];
  const float* b_fa    = (const float*)d_in[16];
  const float* mul_r   = (const float*)d_in[17];
  const float* mul_f   = (const float*)d_in[18];

  float* out = (float*)d_out;
  __bf16* h1 = (__bf16*)d_ws;                                       // 2*M*D bf16
  float* kth = (float*)((char*)d_ws + (size_t)2 * M_ * D_ * 2);     // 2*N*C floats

  dim3 blk(256);
  dim3 g12(M_ / 64, D_ / 256, 2);
  k_gemm1<<<g12, blk, 0, stream>>>(inputs, W_fc_r, b_fc_r, W_fc_f, b_fc_f, h1);
  k_gemm2<<<g12, blk, 0, stream>>>(h1, W_fc1_r, b_fc1_r, W_fc1_f, b_fc1_f, out);
  dim3 g3(M_ / 64, 1, 2);
  k_gemm3<<<g3, blk, 0, stream>>>(out, W_cls_r, b_cls_r, W_ra, b_ra,
                                  W_cls_f, b_cls_f, W_fa, b_fa, out);
  dim3 gk(N_ * C_, 2);
  k_kth<<<gk, blk, 0, stream>>>(out + OFF_CXR, out + OFF_CXF, kth);
  k_final<<<(M_ * C_ + 255) / 256, blk, 0, stream>>>(out, kth, mul_r, mul_f);
}
